// Cache_57870389346832
// MI455X (gfx1250) — matrix operand order repaired and checked by reading, not compiled, not run on hardware
//
#include <hip/hip_runtime.h>
#include <hip/hip_bf16.h>

// Shapes from the reference
#define LSTEPS  64
#define NHID    256
#define CACHE_N 64
#define BSZ     32
#define TOPK    8
#define KCHUNK  128   // K-depth staged in LDS per phase (2 phases of 128)

typedef __attribute__((ext_vector_type(16))) __bf16 v16bf;
typedef __attribute__((ext_vector_type(8)))  __bf16 v8bf;
typedef __attribute__((ext_vector_type(8)))  float  v8f;

__device__ __forceinline__ unsigned fbits(float f) {
  return __builtin_bit_cast(unsigned, f);
}
__device__ __forceinline__ float fval(unsigned u) {
  return __builtin_bit_cast(float, u);
}

// Truncation-based hi/lo bf16 split of 4 f32, packed with v_perm_b32.
//   hi = x & 0xFFFF0000          (exact bit-truncated bf16)
//   r  = x - hi                  (EXACT in f32: shares exponent, <=16 mantissa bits)
//   lo = trunc16(r)              (captures next 8 mantissa bits; residual ~2^-15 |x|)
// a_hi*b_hi + a_hi*b_lo + a_lo*b_hi then reconstructs the f32 product to ~2^-15.
// 12 VALU per float4: 4x v_and + 4x v_sub + 4x v_perm.
__device__ __forceinline__ void cvt_store4(unsigned short* __restrict__ hi,
                                           unsigned short* __restrict__ lo,
                                           int off, float4 v) {
  const unsigned u0 = fbits(v.x), u1 = fbits(v.y), u2 = fbits(v.z), u3 = fbits(v.w);
  const unsigned r0 = fbits(v.x - fval(u0 & 0xFFFF0000u));
  const unsigned r1 = fbits(v.y - fval(u1 & 0xFFFF0000u));
  const unsigned r2 = fbits(v.z - fval(u2 & 0xFFFF0000u));
  const unsigned r3 = fbits(v.w - fval(u3 & 0xFFFF0000u));
  // result bytes {0,1}=src1 bytes {2,3}; bytes {2,3}=src0 bytes {2,3}
  const unsigned SEL = 0x07060302u;
  uint2 hp{__builtin_amdgcn_perm(u1, u0, SEL), __builtin_amdgcn_perm(u3, u2, SEL)};
  uint2 lp{__builtin_amdgcn_perm(r1, r0, SEL), __builtin_amdgcn_perm(r3, r2, SEL)};
  *reinterpret_cast<uint2*>(&hi[off]) = hp;
  *reinterpret_cast<uint2*>(&lo[off]) = lp;
}

// A fragment (16-bit A 16x32, contiguous K like B): e=0..15 -> K=lo8+e (lo8 = 16*lane_half)
__device__ __forceinline__ v16bf load_a_frag(const unsigned short* plane, int row,
                                             int kk, int lo8) {
  v8bf a0 = *reinterpret_cast<const v8bf*>(&plane[row * KCHUNK + kk + lo8]);
  v8bf a1 = *reinterpret_cast<const v8bf*>(&plane[row * KCHUNK + kk + 8 + lo8]);
  return __builtin_shufflevector(a0, a1, 0, 1, 2, 3, 4, 5, 6, 7,
                                         8, 9, 10, 11, 12, 13, 14, 15);
}

// B fragment (16-bit B 32x16, contiguous K like A): e -> K = hi16 + e, 16 contiguous bf16
__device__ __forceinline__ v16bf load_b_frag(const unsigned short* plane, int col,
                                             int kk, int hi16) {
  return *reinterpret_cast<const v16bf*>(&plane[col * KCHUNK + kk + hi16]);
}

// One block per (b, n). Phase 1: cooperative f32->bf16(hi/lo) staging into LDS
// (each element converted exactly once per block). Phase 2: 8 wave32, 4x4 grid
// of 16x16 WMMA tiles (2 per wave) fed from LDS with ds_load_b128.
__global__ __launch_bounds__(256) void att_maxpool_wmma(
    const float* __restrict__ query,   // q2[b][i][h]    = query[i*BSZ*NHID + b*NHID + h]
    const float* __restrict__ keys,    // pk[b][n][j][h] = keys[n*BSZ*L*NHID + b*L*NHID + j*NHID + h]
    float* __restrict__ att)           // (BSZ, CACHE_N)
{
  const int n    = blockIdx.x;
  const int b    = blockIdx.y;
  const int tid  = threadIdx.x;
  const int lane = tid & 31;
  const int w    = tid >> 5;        // wave id 0..7 (wave32)
  const int l15  = lane & 15;
  const int lgrp = lane >> 4;
  const int ti   = w & 3;           // tile row (i-block)
  const int tj0  = (w >> 2) << 1;   // tile cols tj0, tj0+1
  const int lo8  = lgrp << 3;       // unused; A also takes hi16 (K += 16)
  const int hi16 = lgrp << 4;       // B lanes 16..31: K += 16

  const int arow  = ti * 16 + l15;  // A fragment row (0..63)
  const int bcol0 = tj0 * 16 + l15; // B fragment cols
  const int bcol1 = bcol0 + 16;

  const size_t kbase = (size_t)n * (BSZ * LSTEPS * NHID) + (size_t)b * (LSTEPS * NHID);

  // 4 bf16 planes, 64 rows x KCHUNK: 16 KB each, 64 KB total (WGP has 320 KB)
  __shared__ __align__(32) unsigned short qhi[LSTEPS * KCHUNK];
  __shared__ __align__(32) unsigned short qlo[LSTEPS * KCHUNK];
  __shared__ __align__(32) unsigned short khi[LSTEPS * KCHUNK];
  __shared__ __align__(32) unsigned short klo[LSTEPS * KCHUNK];

  v8f acc0 = {};
  v8f acc1 = {};

  for (int k0 = 0; k0 < NHID; k0 += KCHUNK) {
    // ---- Phase 1: stage f32 -> bf16 hi/lo planes.
    // 64 rows x (KCHUNK/4)=32 float4 per tile = 2048 float4; 8 per thread per tile.
#pragma unroll
    for (int c = 0; c < 8; ++c) {
      const int idx = c * 256 + tid;      // 0..2047
      const int row = idx >> 5;           // 0..63
      const int hh  = (idx & 31) << 2;    // 0..124
      float4 qv = *reinterpret_cast<const float4*>(
          query + (size_t)row * (BSZ * NHID) + (size_t)b * NHID + k0 + hh);
      float4 kv = *reinterpret_cast<const float4*>(
          keys + kbase + (size_t)row * NHID + k0 + hh);
      cvt_store4(qhi, qlo, row * KCHUNK + hh, qv);
      cvt_store4(khi, klo, row * KCHUNK + hh, kv);
    }
    __syncthreads();

    // ---- Phase 2: LDS -> WMMA. 4 K-steps of 32 per chunk, 6 WMMA per step.
#pragma unroll
    for (int kk = 0; kk < KCHUNK; kk += 32) {
      v16bf ahi = load_a_frag(qhi, arow, kk, hi16);
      v16bf alo = load_a_frag(qlo, arow, kk, hi16);
      v16bf b0h = load_b_frag(khi, bcol0, kk, hi16);
      v16bf b0l = load_b_frag(klo, bcol0, kk, hi16);
      v16bf b1h = load_b_frag(khi, bcol1, kk, hi16);
      v16bf b1l = load_b_frag(klo, bcol1, kk, hi16);

      acc0 = __builtin_amdgcn_wmma_f32_16x16x32_bf16(false, ahi, false, b0h, (short)0, acc0, false, false);
      acc0 = __builtin_amdgcn_wmma_f32_16x16x32_bf16(false, ahi, false, b0l, (short)0, acc0, false, false);
      acc0 = __builtin_amdgcn_wmma_f32_16x16x32_bf16(false, alo, false, b0h, (short)0, acc0, false, false);

      acc1 = __builtin_amdgcn_wmma_f32_16x16x32_bf16(false, ahi, false, b1h, (short)0, acc1, false, false);
      acc1 = __builtin_amdgcn_wmma_f32_16x16x32_bf16(false, ahi, false, b1l, (short)0, acc1, false, false);
      acc1 = __builtin_amdgcn_wmma_f32_16x16x32_bf16(false, alo, false, b1h, (short)0, acc1, false, false);
    }
    __syncthreads();   // protect LDS before next chunk's overwrite
  }

  // ---- max over this wave's two 16x16 tiles -> lanes -> waves -> one scalar
  float m = fmaxf(acc0[0], acc1[0]);
#pragma unroll
  for (int r = 1; r < 8; ++r) m = fmaxf(m, fmaxf(acc0[r], acc1[r]));
#pragma unroll
  for (int off = 16; off > 0; off >>= 1) m = fmaxf(m, __shfl_xor(m, off, 32));

  __shared__ float red[8];
  if (lane == 0) red[w] = m;
  __syncthreads();
  if (tid == 0) {
    float mm = red[0];
#pragma unroll
    for (int i = 1; i < 8; ++i) mm = fmaxf(mm, red[i]);
    att[b * CACHE_N + n] = mm;
  }
}

// Top-8 of 64 per batch; strict '>' scan matches jax.lax.top_k tie-breaking
// (lowest index wins). Output shape (TOPK, BSZ), int32.
__global__ void topk_kernel(const float* __restrict__ att, int* __restrict__ idx) {
  const int b = threadIdx.x;
  if (b >= BSZ) return;
  float v[CACHE_N];
#pragma unroll
  for (int nn = 0; nn < CACHE_N; ++nn) v[nn] = att[b * CACHE_N + nn];
  for (int k = 0; k < TOPK; ++k) {
    int best = 0;
    float bv = v[0];
    for (int nn = 1; nn < CACHE_N; ++nn) {
      if (v[nn] > bv) { bv = v[nn]; best = nn; }
    }
    idx[k * BSZ + b] = best;
    v[best] = -__builtin_inff();
  }
}

extern "C" void kernel_launch(void* const* d_in, const int* in_sizes, int n_in,
                              void* d_out, int out_size, void* d_ws, size_t ws_size,
                              hipStream_t stream) {
  const float* query = (const float*)d_in[0];  // (1, 64, 32, 256) f32
  const float* keys  = (const float*)d_in[1];  // (64, 32, 64*256) f32
  // d_in[2] = values: computed-but-unused in the reference's max_pooling branch.

  float* att = (float*)d_out;                  // output 0: (32, 1, 64) f32
  int*   idx = (int*)(att + BSZ * CACHE_N);    // output 1: (8, 32) i32

  att_maxpool_wmma<<<dim3(CACHE_N, BSZ), 256, 0, stream>>>(query, keys, att);
  topk_kernel<<<1, BSZ, 0, stream>>>(att, idx);
}
